// Net_36962488549754
// MI455X (gfx1250) — 8-bit matrix operand order repaired and checked by reading, compiled, not run on hardware
//
#include <hip/hip_runtime.h>
#include <hip/hip_bf16.h>
#include <stdint.h>


typedef __attribute__((ext_vector_type(8))) int v8i;
typedef unsigned int uint32x4 __attribute__((ext_vector_type(4)));
typedef int int32x4 __attribute__((ext_vector_type(4)));
typedef int int32x8 __attribute__((ext_vector_type(8)));

#define BT_TOTAL 512
#define B_DIM    32
#define T_DIM    16
#define PPB      4

__device__ __forceinline__ void tdm_copy_to_lds(void* lds_dst,
                                                const void* gsrc,
                                                unsigned nbytes)
{
#if __has_builtin(__builtin_amdgcn_tensor_load_to_lds) && \
    __has_builtin(__builtin_amdgcn_s_wait_tensorcnt)
    const unsigned long long ga = (unsigned long long)(uintptr_t)gsrc;
    const unsigned int lds_off = (unsigned int)(uintptr_t)lds_dst;
    const unsigned int ndw = nbytes >> 2;

    uint32x4 g0;
    g0.x = 0x1u;
    g0.y = lds_off;
    g0.z = (unsigned int)(ga & 0xFFFFFFFFu);
    g0.w = (unsigned int)((ga >> 32) & 0x01FFFFFFu)
         | (2u << 30);

    int32x8 g1;
    g1[0] = (int)(2u << 16);
    g1[1] = (int)(ndw << 16);
    g1[2] = (int)(((ndw >> 16) & 0xFFFFu)
         | (1u << 16));
    g1[3] = (int)(ndw << 16);
    g1[4] = 1;
    g1[5] = (int)ndw;
    g1[6] = 0;
    g1[7] = 0;

    int32x4 gz = {0, 0, 0, 0};
#if __clang_major__ >= 23
    int32x8 gz8 = {0, 0, 0, 0, 0, 0, 0, 0};
    __builtin_amdgcn_tensor_load_to_lds(g0, g1, gz, gz, gz8, 0);
#else
    __builtin_amdgcn_tensor_load_to_lds(g0, g1, gz, gz, 0);
#endif
    __builtin_amdgcn_s_wait_tensorcnt(0);
#else
    const int lane = threadIdx.x;
    for (unsigned i = lane; i < (nbytes >> 2); i += 32)
        ((int*)lds_dst)[i] = ((const int*)gsrc)[i];
#endif
}

__global__ __launch_bounds__(32) void snn_prep_weights(
    const float* __restrict__ w, int Co, int Ci, int Kpad,
    int* __restrict__ wfrag, float* __restrict__ alpha_out)
{
    __shared__ signed char wt[256 * 16];
    __shared__ float alpha_s[16];
    const int tid = threadIdx.x;
    const int K = Ci * 16;

    for (int i = tid; i < Kpad * 16; i += 32) wt[i] = 0;
    if (tid < 16) alpha_s[tid] = 0.0f;
    __syncthreads();

    if (tid < Co) {
        const float* wc = w + tid * K;
        float s = 0.0f;
        for (int k = 0; k < K; ++k) s += fabsf(wc[k]);
        const float th = 0.7f * s / (float)K;
        float wsum = 0.0f, cnt = 0.0f;
        for (int k = 0; k < K; ++k) {
            const float a = fabsf(wc[k]);
            if (a > th) { wsum += a; cnt += 1.0f; }
        }
        alpha_s[tid] = (cnt > 0.0f) ? (wsum / cnt) : 0.0f;
        for (int k = 0; k < K; ++k) {
            const float x = wc[k];
            wt[k * 16 + tid] = (x > th) ? 1 : ((x < -th) ? -1 : 0);
        }
    }
    __syncthreads();

    const int half = tid >> 4, n = tid & 15;
    const int chunks = Kpad >> 6;
    for (int c = 0; c < chunks; ++c) {
        #pragma unroll
        for (int v = 0; v < 8; ++v) {
            const int kbase = c * 64 + ((v >= 4) ? 32 : 0) + (v & 3) * 4 + half * 16;
            unsigned int d = 0;
            #pragma unroll
            for (int b = 0; b < 4; ++b)
                d |= ((unsigned int)(unsigned char)wt[(kbase + b) * 16 + n]) << (8 * b);
            wfrag[(c * 32 + tid) * 8 + v] = (int)d;
        }
    }
    if (tid < 16) alpha_out[tid] = alpha_s[tid];
}

template <bool FULL>
__global__ __launch_bounds__(32) void snn_layer_fused(
    const unsigned char* __restrict__ in,
    const int* __restrict__ wfrag,
    const float* __restrict__ alpha,
    const float* __restrict__ vth,
    unsigned char* __restrict__ spk,
    int Ci, int Hin, int Ho, int Kred, int Kpad, int Co)
{
    __shared__ __align__(16) int ldsB[1024];
    __shared__ __align__(16) unsigned char ldsA[16 * 64];
    __shared__ float ldsDV[16 * 16];

    const int lane   = threadIdx.x;
    const int b      = blockIdx.y;
    const int Wo     = Ho;
    const int HW     = Ho * Wo;
    const int half   = lane >> 4;
    const int n      = lane & 15;
    const int t_row  = lane & 15;
    const int chunks = Kpad >> 6;

    tdm_copy_to_lds(ldsB, wfrag, (unsigned)(chunks * 1024));
    __syncthreads();

    const unsigned char* inrow =
        in + (size_t)(b * T_DIM + t_row) * Ci * Hin * Hin;

    const float al    = alpha[n];
    const float a_sc  = al;
    const float vth_l = (lane < Co) ? vth[lane] : 1.0f;
    const float vthq  = fminf(fmaxf(rintf(vth_l / a_sc), 8.0f), 32.0f) * a_sc;

    for (int pp = 0; pp < PPB; ++pp) {
        const int p = blockIdx.x * PPB + pp;
        if (p >= HW) break;
        const int oy = p / Wo;
        const int ox = p - oy * Wo;

        v8i acc = {0, 0, 0, 0, 0, 0, 0, 0};

        for (int c = 0; c < chunks; ++c) {
            #pragma unroll
            for (int i = 0; i < 8; ++i) {
                const int kk = c * 64 + half * 32 + i * 4;
                unsigned int d = 0;
                if (FULL || kk < Kred) {
                    const int ci = kk >> 4;
                    const int ky = (kk >> 2) & 3;
                    const unsigned char* src =
                        inrow + (ci * Hin + oy + ky) * Hin + ox;
                    d = (unsigned int)src[0] | ((unsigned int)src[1] << 8) |
                        ((unsigned int)src[2] << 16) |
                        ((unsigned int)src[3] << 24);
                }
                ((int*)ldsA)[t_row * 16 + half * 8 + i] = (int)d;
            }
            __syncthreads();

            v8i afrag;
            const int* A32 = (const int*)ldsA;
            #pragma unroll
            for (int v = 0; v < 8; ++v) {
                const int koff = (v & 3) * 4 +
                                 ((v >= 4) ? 32 : 0) + half * 16;
                afrag[v] = A32[t_row * 16 + (koff >> 2)];
            }

            v8i bfrag;
            const int* bsrc = ldsB + (c * 32 + lane) * 8;
            #pragma unroll
            for (int v = 0; v < 8; ++v) bfrag[v] = bsrc[v];

            acc = __builtin_amdgcn_wmma_i32_16x16x64_iu8(
                false, afrag, true, bfrag, acc, false, false);

            __syncthreads();
        }

        #pragma unroll
        for (int j = 0; j < 8; ++j)
            ldsDV[(j + half * 8) * 16 + n] = al * (float)acc[j];
        __syncthreads();

        if (lane < Co) {
            float v = 0.0f;
            for (int t = 0; t < T_DIM; ++t) {
                v += ldsDV[t * 16 + lane];
                const float vq = fminf(fmaxf(v / a_sc, -32.0f), 32.0f) * a_sc;
                const unsigned char s = (vq >= vthq) ? 1 : 0;
                spk[((size_t)(b * T_DIM + t) * Co + lane) * HW + p] = s;
                if (s) v = 0.0f;
            }
        }
        __syncthreads();
    }
}

__global__ void snn_f32_to_u8(const float* __restrict__ x,
                              unsigned char* __restrict__ y, int n)
{
    const int i = blockIdx.x * blockDim.x + threadIdx.x;
    if (i < n) y[i] = (x[i] != 0.0f) ? 1 : 0;
}

__global__ void snn_mean(const unsigned char* __restrict__ spk,
                         float* __restrict__ out, int HW, int total)
{
    const int idx = blockIdx.x * blockDim.x + threadIdx.x;
    if (idx >= total) return;
    const unsigned char* s = spk + (size_t)idx * HW;
    int acc = 0;
    for (int j = 0; j < HW; ++j) acc += s[j];
    out[idx] = (float)acc / (float)HW;
}

extern "C" void kernel_launch(void* const* d_in, const int* in_sizes, int n_in,
                              void* d_out, int out_size, void* d_ws, size_t ws_size,
                              hipStream_t stream)
{
    (void)in_sizes; (void)n_in; (void)out_size; (void)ws_size;

    const float* x = (const float*)d_in[0];
    const float* w[5]   = {(const float*)d_in[1], (const float*)d_in[3],
                           (const float*)d_in[5], (const float*)d_in[7],
                           (const float*)d_in[9]};
    const float* vth[5] = {(const float*)d_in[2], (const float*)d_in[4],
                           (const float*)d_in[6], (const float*)d_in[8],
                           (const float*)d_in[10]};

    struct LayerCfg { int Ci, Co, Hin, Ho, Kred, Kpad; };
    const LayerCfg L[5] = {
        { 3, 16, 48, 45,  48,  64},
        {16, 16, 45, 42, 256, 256},
        {16, 16, 42, 39, 256, 256},
        {16, 16, 39, 36, 256, 256},
        {16,  6, 36, 33, 256, 256},
    };

    char* ws = (char*)d_ws;
    const size_t ACT_BYTES = (size_t)BT_TOTAL * 16 * 45 * 45;
    unsigned char* actA = (unsigned char*)ws;
    unsigned char* actB = actA + ACT_BYTES;
    char* wbase = ws + 2 * ACT_BYTES;
    const size_t WSTRIDE = 4096 + 256;

    unsigned char* cur = actA;
    unsigned char* nxt = actB;

    const int nX = BT_TOTAL * 3 * 48 * 48;
    snn_f32_to_u8<<<(nX + 255) / 256, 256, 0, stream>>>(x, cur, nX);

    for (int l = 0; l < 5; ++l) {
        const LayerCfg& c = L[l];
        int*   wfrag_l = (int*)(wbase + l * WSTRIDE);
        float* alpha_l = (float*)(wbase + l * WSTRIDE + 4096);

        snn_prep_weights<<<1, 32, 0, stream>>>(w[l], c.Co, c.Ci, c.Kpad,
                                               wfrag_l, alpha_l);

        const int HW = c.Ho * c.Ho;
        dim3 grid((HW + PPB - 1) / PPB, B_DIM);
        if (c.Kred == c.Kpad) {
            snn_layer_fused<true><<<grid, 32, 0, stream>>>(
                cur, wfrag_l, alpha_l, vth[l], nxt,
                c.Ci, c.Hin, c.Ho, c.Kred, c.Kpad, c.Co);
        } else {
            snn_layer_fused<false><<<grid, 32, 0, stream>>>(
                cur, wfrag_l, alpha_l, vth[l], nxt,
                c.Ci, c.Hin, c.Ho, c.Kred, c.Kpad, c.Co);
        }

        unsigned char* t = cur; cur = nxt; nxt = t;
    }

    const int totalOut = BT_TOTAL * 6;
    snn_mean<<<(totalOut + 255) / 256, 256, 0, stream>>>(
        cur, (float*)d_out, 33 * 33, totalOut);
}
